// GAT_18004502905471
// MI455X (gfx1250) — matrix operand order repaired and checked by reading, not compiled, not run on hardware
//
#include <hip/hip_runtime.h>
#include <hip/hip_bf16.h>

#define HEADS 8
#define HID 32
#define C1 (HEADS * HID)   /* 256 */
#define NCLS 40
#define NPAD 48
#define NEG_SLOPE 0.2f
#define EPSV 1e-16f

typedef __attribute__((ext_vector_type(16))) __bf16 v16bf;
typedef __attribute__((ext_vector_type(8)))  float  v8f;

// ---------- helpers ----------
__device__ __forceinline__ unsigned short f2bf(float f) {   // RNE f32 -> bf16
  unsigned u = __float_as_uint(f);
  u += 0x7FFFu + ((u >> 16) & 1u);
  return (unsigned short)(u >> 16);
}
__device__ __forceinline__ unsigned f2ord(float f) {        // order-preserving map
  unsigned u = __float_as_uint(f);
  return (u & 0x80000000u) ? ~u : (u | 0x80000000u);
}
__device__ __forceinline__ float ord2f(unsigned m) {
  unsigned u = (m & 0x80000000u) ? (m & 0x7FFFFFFFu) : ~m;
  return __uint_as_float(u);
}
__device__ __forceinline__ float lrelu(float x) { return x > 0.f ? x : NEG_SLOPE * x; }

// ---------- elementwise ----------
__global__ void k_f32_to_bf16(const float* __restrict__ in, unsigned short* __restrict__ out, size_t n) {
  size_t i = (size_t)blockIdx.x * blockDim.x + threadIdx.x;
  if (i < n) out[i] = f2bf(in[i]);
}
__global__ void k_zero_u32(unsigned* __restrict__ p, size_t n) {
  size_t i = (size_t)blockIdx.x * blockDim.x + threadIdx.x;
  if (i < n) p[i] = 0u;
}
__global__ void k_init_bias(float* __restrict__ out, const float* __restrict__ bias, size_t n, int C) {
  size_t i = (size_t)blockIdx.x * blockDim.x + threadIdx.x;
  if (i < n) out[i] = bias[i % C];
}
__global__ void k_elu_bf16(const float* __restrict__ in, unsigned short* __restrict__ out, size_t n) {
  size_t i = (size_t)blockIdx.x * blockDim.x + threadIdx.x;
  if (i >= n) return;
  float x = in[i];
  out[i] = f2bf(x > 0.f ? x : (expf(x) - 1.f));
}

// ---------- pack W (f32, K x Nin row-major) into per-lane WMMA B-fragment order ----------
// Fragment layout (16-bit B 32x16, contiguous k order used by this code): lane L -> column n = L&15,
// K range [kk + 16*(L>>4), +16). Packed element (kt, nt, L, e) stored contiguously:
//   out[((kt*ntiles + nt)*32 + L)*16 + e] = B[kt*32 + 16*(L>>4) + e][nt*16 + (L&15)]
__global__ void k_pack_w(const float* __restrict__ W, unsigned short* __restrict__ out,
                         int K, int Nin, int Npad) {
  int ntiles = Npad >> 4;
  int total = (K >> 5) * ntiles * 512;
  int idx = blockIdx.x * blockDim.x + threadIdx.x;
  if (idx >= total) return;
  int e = idx & 15;
  int L = (idx >> 4) & 31;
  int tile = idx >> 9;
  int nt = tile % ntiles, kt = tile / ntiles;
  int n = nt * 16 + (L & 15);
  int k = kt * 32 + 16 * (L >> 4) + e;
  out[idx] = (n < Nin) ? f2bf(W[(size_t)k * Nin + n]) : (unsigned short)0;
}

// ---------- WMMA bf16 GEMM: C[M,N] = A[M,K] * Bpacked, one wave per 16 x (16*NT) tile ----------
template <int NT>
__global__ void __launch_bounds__(32)
k_gemm_bf16(const unsigned short* __restrict__ A, const unsigned short* __restrict__ Bp,
            float* __restrict__ C, int M, int N, int K) {
  const int lane = threadIdx.x;
  const int g = lane >> 4, r = lane & 15;
  const int nTiles = N >> 4;
  const int nt0 = blockIdx.x * NT;
  const int m0 = blockIdx.y * 16;
  int rowA = m0 + r; if (rowA > M - 1) rowA = M - 1;

  v8f acc[NT];
#pragma unroll
  for (int t = 0; t < NT; ++t) acc[t] = (v8f){};

  for (int kk = 0; kk < K; kk += 32) {
    // A fragment (16x32 bf16): lane covers K in [kk+16g, +16)
    union { v16bf v; uint4 q[2]; } af;
    const unsigned short* arow = A + (size_t)rowA * K + kk + 16 * g;
    af.q[0] = *(const uint4*)(arow);
    af.q[1] = *(const uint4*)(arow + 8);
    const int kt = kk >> 5;
#pragma unroll
    for (int t = 0; t < NT; ++t) {
      union { v16bf v; uint4 q[2]; } bfr;
      const uint4* bq = (const uint4*)(Bp + ((size_t)(kt * nTiles + nt0 + t) * 32 + lane) * 16);
      bfr.q[0] = bq[0];
      bfr.q[1] = bq[1];
      acc[t] = __builtin_amdgcn_wmma_f32_16x16x32_bf16(false, af.v, false, bfr.v,
                                                       (short)0, acc[t], false, false);
    }
  }
#pragma unroll
  for (int t = 0; t < NT; ++t) {
    int n0 = (nt0 + t) * 16;
#pragma unroll
    for (int i = 0; i < 8; ++i) {
      int row = m0 + i + 8 * g;
      if (row < M) C[(size_t)row * N + n0 + r] = acc[t][i];
    }
  }
}

// ---------- per-node attention halves: al[n,h] = sum_d H[n, h*dim+d] * a[h*dim+d] ----------
__global__ void k_al(const float* __restrict__ H, const float* __restrict__ asrc,
                     const float* __restrict__ adst, float* __restrict__ osrc,
                     float* __restrict__ odst, int Nn, int heads, int dim, int stride) {
  int i = blockIdx.x * blockDim.x + threadIdx.x;
  if (i >= Nn * heads) return;
  int n = i / heads, h = i % heads;
  const float* base = H + (size_t)n * stride + h * dim;
  float s = 0.f, t = 0.f;
  for (int d = 0; d < dim; ++d) {
    float v = base[d];
    s += v * asrc[h * dim + d];
    t += v * adst[h * dim + d];
  }
  osrc[i] = s; odst[i] = t;
}

// ---------- edge passes ----------
__global__ void k_edge_max(const int* __restrict__ ei, int Eb, int Et,
                           const float* __restrict__ s, const float* __restrict__ t,
                           unsigned* __restrict__ m, int H) {
  int e = blockIdx.x * blockDim.x + threadIdx.x;
  if (e >= Et) return;
  int si = (e < Eb) ? ei[e] : e - Eb;
  int di = (e < Eb) ? ei[Eb + e] : e - Eb;
  for (int h = 0; h < H; ++h) {
    float lg = lrelu(s[si * H + h] + t[di * H + h]);
    atomicMax(&m[di * H + h], f2ord(lg));
  }
}
__global__ void k_edge_denom(const int* __restrict__ ei, int Eb, int Et,
                             const float* __restrict__ s, const float* __restrict__ t,
                             const unsigned* __restrict__ m, float* __restrict__ den, int H) {
  int e = blockIdx.x * blockDim.x + threadIdx.x;
  if (e >= Et) return;
  int si = (e < Eb) ? ei[e] : e - Eb;
  int di = (e < Eb) ? ei[Eb + e] : e - Eb;
  for (int h = 0; h < H; ++h) {
    float lg = lrelu(s[si * H + h] + t[di * H + h]);
    atomicAdd(&den[di * H + h], expf(lg - ord2f(m[di * H + h])));
  }
}

// layer-1 weighted scatter: block per edge, 256 threads (one per h*32+d channel)
__global__ void __launch_bounds__(C1)
k_scatter1(const int* __restrict__ ei, int Eb,
           const float* __restrict__ s1, const float* __restrict__ t1,
           const unsigned* __restrict__ m1, const float* __restrict__ den1,
           const float* __restrict__ H1, float* __restrict__ out1) {
  int e = blockIdx.x;
  int si = (e < Eb) ? ei[e] : e - Eb;
  int di = (e < Eb) ? ei[Eb + e] : e - Eb;
  __shared__ float sa[HEADS];
  int c = threadIdx.x;
  if (c < HEADS) {
    float lg = lrelu(s1[si * HEADS + c] + t1[di * HEADS + c]);
    float ex = expf(lg - ord2f(m1[di * HEADS + c]));
    sa[c] = ex / (den1[di * HEADS + c] + EPSV);
  }
  __syncthreads();
  float v = H1[(size_t)si * C1 + c] * sa[c >> 5];
  atomicAdd(&out1[(size_t)di * C1 + c], v);
}

// layer-2 weighted scatter: block per edge, 64 threads (40 active)
__global__ void __launch_bounds__(64)
k_scatter2(const int* __restrict__ ei, int Eb,
           const float* __restrict__ s2, const float* __restrict__ t2,
           const unsigned* __restrict__ m2, const float* __restrict__ den2,
           const float* __restrict__ H2, float* __restrict__ out) {
  int e = blockIdx.x;
  int si = (e < Eb) ? ei[e] : e - Eb;
  int di = (e < Eb) ? ei[Eb + e] : e - Eb;
  __shared__ float sa;
  if (threadIdx.x == 0) {
    float lg = lrelu(s2[si] + t2[di]);
    float ex = expf(lg - ord2f(m2[di]));
    sa = ex / (den2[di] + EPSV);
  }
  __syncthreads();
  int c = threadIdx.x;
  if (c < NCLS)
    atomicAdd(&out[(size_t)di * NCLS + c], H2[(size_t)si * NPAD + c] * sa);
}

__global__ void k_log_softmax(float* __restrict__ out, int Nn, int C) {
  int n = blockIdx.x * blockDim.x + threadIdx.x;
  if (n >= Nn) return;
  float* row = out + (size_t)n * C;
  float mx = -3.4e38f;
  for (int c = 0; c < C; ++c) mx = fmaxf(mx, row[c]);
  float sum = 0.f;
  for (int c = 0; c < C; ++c) sum += expf(row[c] - mx);
  float lse = mx + logf(sum);
  for (int c = 0; c < C; ++c) row[c] -= lse;
}

extern "C" void kernel_launch(void* const* d_in, const int* in_sizes, int n_in,
                              void* d_out, int out_size, void* d_ws, size_t ws_size,
                              hipStream_t stream) {
  const float* x   = (const float*)d_in[0];
  const int*   ei  = (const int*)d_in[1];
  const float* W1  = (const float*)d_in[2];
  const float* as1 = (const float*)d_in[3];
  const float* ad1 = (const float*)d_in[4];
  const float* b1  = (const float*)d_in[5];
  const float* W2  = (const float*)d_in[6];
  const float* as2 = (const float*)d_in[7];
  const float* ad2 = (const float*)d_in[8];
  const float* b2  = (const float*)d_in[9];
  float* out = (float*)d_out;

  const int IN = 128;
  const int N  = in_sizes[0] / IN;       // 50000
  const int Eb = in_sizes[1] / 2;        // 800000
  const int Et = Eb + N;                 // + self loops
  (void)n_in; (void)out_size; (void)ws_size;

  // workspace carve-out
  char* w = (char*)d_ws; size_t off = 0;
  auto alloc = [&](size_t b) -> void* {
    void* p = w + off; off = (off + b + 255) & ~(size_t)255; return p;
  };
  unsigned short* Xbf  = (unsigned short*)alloc((size_t)N * IN * 2);
  unsigned short* W1p  = (unsigned short*)alloc((size_t)IN * C1 * 2);    // packed fragments
  float*    H1   = (float*)   alloc((size_t)N * C1 * 4);
  float*    s1   = (float*)   alloc((size_t)N * HEADS * 4);
  float*    t1   = (float*)   alloc((size_t)N * HEADS * 4);
  unsigned* m1   = (unsigned*)alloc((size_t)N * HEADS * 4);
  float*    den1 = (float*)   alloc((size_t)N * HEADS * 4);
  float*    out1 = (float*)   alloc((size_t)N * C1 * 4);
  unsigned short* A2bf = (unsigned short*)alloc((size_t)N * C1 * 2);
  unsigned short* W2p  = (unsigned short*)alloc((size_t)C1 * NPAD * 2);  // packed fragments
  float*    H2   = (float*)   alloc((size_t)N * NPAD * 4);
  float*    s2   = (float*)   alloc((size_t)N * 4);
  float*    t2   = (float*)   alloc((size_t)N * 4);
  unsigned* m2   = (unsigned*)alloc((size_t)N * 4);
  float*    den2 = (float*)   alloc((size_t)N * 4);

  auto cdiv = [](long a, long b) { return (unsigned)((a + b - 1) / b); };

  // ---- layer 1 ----
  k_f32_to_bf16<<<cdiv((long)N * IN, 256), 256, 0, stream>>>(x, Xbf, (size_t)N * IN);
  k_pack_w<<<cdiv((long)IN * C1, 256), 256, 0, stream>>>(W1, W1p, IN, C1, C1);
  k_pack_w<<<cdiv((long)C1 * NPAD, 256), 256, 0, stream>>>(W2, W2p, C1, NCLS, NPAD);

  dim3 g1(C1 / 64, (N + 15) / 16);                       // NT=4 -> 4 N-tiles per wave
  k_gemm_bf16<4><<<g1, 32, 0, stream>>>(Xbf, W1p, H1, N, C1, IN);

  k_al<<<cdiv((long)N * HEADS, 256), 256, 0, stream>>>(H1, as1, ad1, s1, t1, N, HEADS, HID, C1);

  k_zero_u32<<<cdiv((long)N * HEADS, 256), 256, 0, stream>>>(m1, (size_t)N * HEADS);
  k_zero_u32<<<cdiv((long)N * HEADS, 256), 256, 0, stream>>>((unsigned*)den1, (size_t)N * HEADS);
  k_edge_max  <<<cdiv(Et, 256), 256, 0, stream>>>(ei, Eb, Et, s1, t1, m1, HEADS);
  k_edge_denom<<<cdiv(Et, 256), 256, 0, stream>>>(ei, Eb, Et, s1, t1, m1, den1, HEADS);

  k_init_bias<<<cdiv((long)N * C1, 256), 256, 0, stream>>>(out1, b1, (size_t)N * C1, C1);
  k_scatter1<<<Et, C1, 0, stream>>>(ei, Eb, s1, t1, m1, den1, H1, out1);

  // ---- ELU + layer 2 ----
  k_elu_bf16<<<cdiv((long)N * C1, 256), 256, 0, stream>>>(out1, A2bf, (size_t)N * C1);

  dim3 g2(1, (N + 15) / 16);                             // NT=3 -> all 48 cols per wave
  k_gemm_bf16<3><<<g2, 32, 0, stream>>>(A2bf, W2p, H2, N, NPAD, C1);

  k_al<<<cdiv(N, 256), 256, 0, stream>>>(H2, as2, ad2, s2, t2, N, 1, NCLS, NPAD);

  k_zero_u32<<<cdiv(N, 256), 256, 0, stream>>>(m2, (size_t)N);
  k_zero_u32<<<cdiv(N, 256), 256, 0, stream>>>((unsigned*)den2, (size_t)N);
  k_edge_max  <<<cdiv(Et, 256), 256, 0, stream>>>(ei, Eb, Et, s2, t2, m2, 1);
  k_edge_denom<<<cdiv(Et, 256), 256, 0, stream>>>(ei, Eb, Et, s2, t2, m2, den2, 1);

  k_init_bias<<<cdiv((long)N * NCLS, 256), 256, 0, stream>>>(out, b2, (size_t)N * NCLS, NCLS);
  k_scatter2<<<Et, 64, 0, stream>>>(ei, Eb, s2, t2, m2, den2, H2, out);

  k_log_softmax<<<cdiv(N, 256), 256, 0, stream>>>(out, N, NCLS);
}
